// HGNN_67559835566710
// MI455X (gfx1250) — matrix operand order repaired and checked by reading, not compiled, not run on hardware
//
#include <hip/hip_runtime.h>
#include <hip/hip_bf16.h>

// ---------------- problem constants (match reference) ----------------
#define CN0   100000
#define CE0   100000
#define CNNZ0 800000
#define CN1   25000
#define CE1   25000
#define CNNZ1 200000
#define CN2   6250
#define CE2   6250
#define CNNZ2 50000
#define IN_DIM 128
#define HID    256
#define OUT_DIM 64

typedef __attribute__((ext_vector_type(16))) _Float16 v16h;
typedef __attribute__((ext_vector_type(8)))  float    v8f;

// ---------------- elementwise / utility kernels ----------------
__global__ __launch_bounds__(256) void zero_k(float4* __restrict__ p, long n4) {
    long i = (long)blockIdx.x * blockDim.x + threadIdx.x;
    long st = (long)gridDim.x * blockDim.x;
    const float4 z = make_float4(0.f, 0.f, 0.f, 0.f);
    for (; i < n4; i += st) p[i] = z;
}

__global__ __launch_bounds__(256) void deg_k(const int* __restrict__ idx, float* __restrict__ out, long n) {
    long i = (long)blockIdx.x * blockDim.x + threadIdx.x;
    long st = (long)gridDim.x * blockDim.x;
    for (; i < n; i += st) unsafeAtomicAdd(&out[idx[i]], 1.0f);
}

__global__ __launch_bounds__(256) void invsqrt_k(float* __restrict__ p, long n) {
    long i = (long)blockIdx.x * blockDim.x + threadIdx.x;
    long st = (long)gridDim.x * blockDim.x;
    for (; i < n; i += st) p[i] = rsqrtf(p[i]);
}

__global__ __launch_bounds__(256) void recip_k(float* __restrict__ p, long n) {
    long i = (long)blockIdx.x * blockDim.x + threadIdx.x;
    long st = (long)gridDim.x * blockDim.x;
    for (; i < n; i += st) p[i] = 1.0f / p[i];
}

// row-scale (by precomputed per-row factor) + optional ReLU, in place; D known at compile time
template <int D, int RELU>
__global__ __launch_bounds__(256) void scale_rows_k(float* __restrict__ Y, const float* __restrict__ s, long n) {
    constexpr int DV = D / 4;                 // power of two -> shifts
    long total = n * (long)DV;
    long i = (long)blockIdx.x * blockDim.x + threadIdx.x;
    long st = (long)gridDim.x * blockDim.x;
    for (; i < total; i += st) {
        long r = i / DV;
        float sc = s[r];
        float4* p = reinterpret_cast<float4*>(Y) + i;
        float4 v = *p;
        v.x *= sc; v.y *= sc; v.z *= sc; v.w *= sc;
        if (RELU) {
            v.x = v.x > 0.f ? v.x : 0.f;
            v.y = v.y > 0.f ? v.y : 0.f;
            v.z = v.z > 0.f ? v.z : 0.f;
            v.w = v.w > 0.f ? v.w : 0.f;
        }
        *p = v;
    }
}

// E[edge[p]] += X[node[p]] * dvi[node[p]]   (one thread = one pin x 4 features)
template <int D>
__global__ __launch_bounds__(256) void lap_edge_k(const float* __restrict__ X,
                                                  const int* __restrict__ node, const int* __restrict__ edge,
                                                  const float* __restrict__ dvi,
                                                  float* __restrict__ E, long nnz) {
    constexpr int DV = D / 4;
    long total = nnz * (long)DV;
    long i = (long)blockIdx.x * blockDim.x + threadIdx.x;
    long st = (long)gridDim.x * blockDim.x;
    for (; i < total; i += st) {
        long p = i / DV;
        int  j4 = (int)(i - p * (long)DV) * 4;
        int  nd = node[p];
        int  ed = edge[p];
        float sc = dvi[nd];
        float4 v = *reinterpret_cast<const float4*>(X + (long)nd * D + j4);
        float* dst = E + (long)ed * D + j4;
        unsafeAtomicAdd(dst + 0, v.x * sc);
        unsafeAtomicAdd(dst + 1, v.y * sc);
        unsafeAtomicAdd(dst + 2, v.z * sc);
        unsafeAtomicAdd(dst + 3, v.w * sc);
    }
}

// Y[node[p]] += E[edge[p]] * dei[edge[p]]
template <int D>
__global__ __launch_bounds__(256) void lap_node_k(const float* __restrict__ E,
                                                  const int* __restrict__ node, const int* __restrict__ edge,
                                                  const float* __restrict__ dei,
                                                  float* __restrict__ Y, long nnz) {
    constexpr int DV = D / 4;
    long total = nnz * (long)DV;
    long i = (long)blockIdx.x * blockDim.x + threadIdx.x;
    long st = (long)gridDim.x * blockDim.x;
    for (; i < total; i += st) {
        long p = i / DV;
        int  j4 = (int)(i - p * (long)DV) * 4;
        int  nd = node[p];
        int  ed = edge[p];
        float sc = dei[ed];
        float4 v = *reinterpret_cast<const float4*>(E + (long)ed * D + j4);
        float* dst = Y + (long)nd * D + j4;
        unsafeAtomicAdd(dst + 0, v.x * sc);
        unsafeAtomicAdd(dst + 1, v.y * sc);
        unsafeAtomicAdd(dst + 2, v.z * sc);
        unsafeAtomicAdd(dst + 3, v.w * sc);
    }
}

// P[assign[r]] += X[r]  (segment-sum for mean pooling)
template <int D>
__global__ __launch_bounds__(256) void pool_k(const float* __restrict__ X, const int* __restrict__ assign,
                                              float* __restrict__ P, long nf) {
    constexpr int DV = D / 4;
    long total = nf * (long)DV;
    long i = (long)blockIdx.x * blockDim.x + threadIdx.x;
    long st = (long)gridDim.x * blockDim.x;
    for (; i < total; i += st) {
        long r = i / DV;
        int  j4 = (int)(i - r * (long)DV) * 4;
        float4 v = *reinterpret_cast<const float4*>(X + r * (long)D + j4);
        float* dst = P + (long)assign[r] * D + j4;
        unsafeAtomicAdd(dst + 0, v.x);
        unsafeAtomicAdd(dst + 1, v.y);
        unsafeAtomicAdd(dst + 2, v.z);
        unsafeAtomicAdd(dst + 3, v.w);
    }
}

// U[r] = concat(Y2[assign[r]], h1[r])  -> (n1, 2D)
template <int D>
__global__ __launch_bounds__(256) void unpool_concat_k(const float* __restrict__ Y2, const float* __restrict__ h1,
                                                       const int* __restrict__ assign,
                                                       float* __restrict__ U, long n1) {
    constexpr int DV = D / 4;
    long total = n1 * (long)DV;
    long i = (long)blockIdx.x * blockDim.x + threadIdx.x;
    long st = (long)gridDim.x * blockDim.x;
    for (; i < total; i += st) {
        long r = i / DV;
        int  j4 = (int)(i - r * (long)DV) * 4;
        float4 a = *reinterpret_cast<const float4*>(Y2 + (long)assign[r] * D + j4);
        float4 b = *reinterpret_cast<const float4*>(h1 + r * (long)D + j4);
        *reinterpret_cast<float4*>(U + r * (long)(2 * D) + j4)     = a;
        *reinterpret_cast<float4*>(U + r * (long)(2 * D) + D + j4) = b;
    }
}

// ---------------- WMMA GEMM: Y(n,dout) = X(n,k) @ W(dout,k)^T + bias ----------------
// One wave32 per 16x16 output tile; K-loop of 32 using v_wmma_f32_16x16x32_f16.
// Fragment layouts per CDNA5 ISA 7.12.2 (16-bit A 16x32, B 32x16, f32 C/D 16x16).
__global__ __launch_bounds__(256) void gemm_bias_wmma_k(const float* __restrict__ X,
                                                        const float* __restrict__ W,
                                                        const float* __restrict__ bias,
                                                        float* __restrict__ Y,
                                                        int n, int k, int dout) {
    const int wave = threadIdx.x >> 5;
    const int lane = threadIdx.x & 31;
    const int tiles_n = dout >> 4;
    const int tiles_m = (n + 15) >> 4;
    const int tile = blockIdx.x * 8 + wave;
    if (tile >= tiles_m * tiles_n) return;     // wave-uniform: EXEC stays all-1s for WMMA
    const int tm = tile / tiles_n;
    const int tn = tile % tiles_n;

    const int half = lane >> 4;   // 0: lanes 0-15, 1: lanes 16-31
    const int r    = lane & 15;

    int arow = tm * 16 + r;
    if (arow >= n) arow = n - 1;               // clamped load; store is masked below
    const int bcol = tn * 16 + r;              // dout is a multiple of 16

    const float* aptr = X + (long)arow * k;
    const float* bptr = W + (long)bcol * k;

    v8f acc = {};
    for (int kb = 0; kb < k; kb += 32) {
        // A fragment: half 0 holds K = kb+{0..7, 16..23}; half 1 holds K = kb+{8..15, 24..31}
        const float4 a0 = *reinterpret_cast<const float4*>(aptr + kb + half * 8);
        const float4 a1 = *reinterpret_cast<const float4*>(aptr + kb + half * 8 + 4);
        const float4 a2 = *reinterpret_cast<const float4*>(aptr + kb + half * 8 + 16);
        const float4 a3 = *reinterpret_cast<const float4*>(aptr + kb + half * 8 + 20);
        // B fragment: lane = output column; half 0 holds K = kb+{0..7, 16..23}; half 1 holds K = kb+{8..15, 24..31}
        const float4 b0 = *reinterpret_cast<const float4*>(bptr + kb + half * 8);
        const float4 b1 = *reinterpret_cast<const float4*>(bptr + kb + half * 8 + 4);
        const float4 b2 = *reinterpret_cast<const float4*>(bptr + kb + half * 8 + 16);
        const float4 b3 = *reinterpret_cast<const float4*>(bptr + kb + half * 8 + 20);

        v16h a, b;
        a[0]=(_Float16)a0.x; a[1]=(_Float16)a0.y; a[2]=(_Float16)a0.z; a[3]=(_Float16)a0.w;
        a[4]=(_Float16)a1.x; a[5]=(_Float16)a1.y; a[6]=(_Float16)a1.z; a[7]=(_Float16)a1.w;
        a[8]=(_Float16)a2.x; a[9]=(_Float16)a2.y; a[10]=(_Float16)a2.z; a[11]=(_Float16)a2.w;
        a[12]=(_Float16)a3.x; a[13]=(_Float16)a3.y; a[14]=(_Float16)a3.z; a[15]=(_Float16)a3.w;
        b[0]=(_Float16)b0.x; b[1]=(_Float16)b0.y; b[2]=(_Float16)b0.z; b[3]=(_Float16)b0.w;
        b[4]=(_Float16)b1.x; b[5]=(_Float16)b1.y; b[6]=(_Float16)b1.z; b[7]=(_Float16)b1.w;
        b[8]=(_Float16)b2.x; b[9]=(_Float16)b2.y; b[10]=(_Float16)b2.z; b[11]=(_Float16)b2.w;
        b[12]=(_Float16)b3.x; b[13]=(_Float16)b3.y; b[14]=(_Float16)b3.z; b[15]=(_Float16)b3.w;

        acc = __builtin_amdgcn_wmma_f32_16x16x32_f16(
            /*neg_a=*/false, a, /*neg_b=*/false, b,
            /*c_mod=*/(short)0, acc, /*reuse_a=*/false, /*reuse_b=*/false);
    }

    // C/D layout: VGPR i -> M = half*8 + i, N = lane&15
    const float bv = bias[bcol];
    float* yp = Y + ((long)(tm * 16 + half * 8)) * dout + bcol;
    if (tm * 16 + 16 <= n) {                   // full tile: wave-uniform fast path, no predication
        #pragma unroll
        for (int i = 0; i < 8; ++i) yp[(long)i * dout] = acc[i] + bv;
    } else {
        #pragma unroll
        for (int i = 0; i < 8; ++i) {
            int row = tm * 16 + half * 8 + i;
            if (row < n) Y[(long)row * dout + bcol] = acc[i] + bv;
        }
    }
}

// ---------------- host-side orchestration ----------------
static inline int nb(long n, int t) {
    long b = (n + t - 1) / t;
    if (b > 1048576L) b = 1048576L;
    if (b < 1) b = 1;
    return (int)b;
}

extern "C" void kernel_launch(void* const* d_in, const int* in_sizes, int n_in,
                              void* d_out, int out_size, void* d_ws, size_t ws_size,
                              hipStream_t stream) {
    (void)in_sizes; (void)n_in; (void)out_size; (void)ws_size;
    const float* X  = (const float*)d_in[0];
    const float* W0 = (const float*)d_in[1];
    const float* b0 = (const float*)d_in[2];
    const float* W1 = (const float*)d_in[3];
    const float* b1 = (const float*)d_in[4];
    const float* W2 = (const float*)d_in[5];
    const float* b2 = (const float*)d_in[6];
    const float* W3 = (const float*)d_in[7];
    const float* b3 = (const float*)d_in[8];
    const float* W4 = (const float*)d_in[9];
    const float* b4 = (const float*)d_in[10];
    const int* node0 = (const int*)d_in[11];
    const int* edge0 = (const int*)d_in[12];
    const int* node1 = (const int*)d_in[13];
    const int* edge1 = (const int*)d_in[14];
    const int* node2 = (const int*)d_in[15];
    const int* edge2 = (const int*)d_in[16];
    const int* assign0 = (const int*)d_in[17];
    const int* assign1 = (const int*)d_in[18];
    float* out = (float*)d_out;

    // ---- carve workspace (floats) ----
    float* ws = (float*)d_ws;
    size_t off = 0;
    auto carve = [&](size_t nf) { float* p = ws + off; off += nf; return p; };
    float* big0 = carve((size_t)CN0 * HID);   // G0 / G1 / U(concat) / G4 ...
    float* big1 = carve((size_t)CE0 * HID);   // E-buffers / G3 ...
    float* big2 = carve((size_t)CN0 * HID);   // Y0 / E3 ...
    float* P1   = carve((size_t)CN1 * HID);
    float* H1   = carve((size_t)CN1 * HID);   // skip connection (persists)
    float* Y3   = carve((size_t)CN1 * HID);
    float* P2   = carve((size_t)CN2 * HID);
    float* G2b  = carve((size_t)CN2 * HID);
    float* E2b  = carve((size_t)CE2 * HID);
    float* Y2   = carve((size_t)CN2 * HID);
    float* dv0  = carve(CN0);
    float* de0  = carve(CE0);
    float* dv1  = carve(CN1);
    float* de1  = carve(CE1);
    float* dv2  = carve(CN2);
    float* de2  = carve(CE2);
    float* c0   = carve(CN1);
    float* c1   = carve(CN2);

    const int T = 256;
    const long degTotal = (long)CN0 + CE0 + CN1 + CE1 + CN2 + CE2 + CN1 + CN2;

    auto zero = [&](float* p, long n) {  // zeroes 4*(n/4) floats; an n % 4 tail is left as is
        zero_k<<<nb(n / 4, T), T, 0, stream>>>(reinterpret_cast<float4*>(p), n / 4);
    };

    // ---- degrees ----
    zero(dv0, degTotal);                       // all degree arrays are contiguous; degTotal % 4 == 2, so the last 2 floats are not zeroed
    deg_k<<<nb(CNNZ0, T), T, 0, stream>>>(node0, dv0, CNNZ0);
    deg_k<<<nb(CNNZ0, T), T, 0, stream>>>(edge0, de0, CNNZ0);
    deg_k<<<nb(CNNZ1, T), T, 0, stream>>>(node1, dv1, CNNZ1);
    deg_k<<<nb(CNNZ1, T), T, 0, stream>>>(edge1, de1, CNNZ1);
    deg_k<<<nb(CNNZ2, T), T, 0, stream>>>(node2, dv2, CNNZ2);
    deg_k<<<nb(CNNZ2, T), T, 0, stream>>>(edge2, de2, CNNZ2);
    deg_k<<<nb(CN0, T), T, 0, stream>>>(assign0, c0, CN0);
    deg_k<<<nb(CN1, T), T, 0, stream>>>(assign1, c1, CN1);
    invsqrt_k<<<nb(CN0, T), T, 0, stream>>>(dv0, CN0);
    invsqrt_k<<<nb(CN1, T), T, 0, stream>>>(dv1, CN1);
    invsqrt_k<<<nb(CN2, T), T, 0, stream>>>(dv2, CN2);
    recip_k<<<nb(CE0, T), T, 0, stream>>>(de0, CE0);
    recip_k<<<nb(CE1, T), T, 0, stream>>>(de1, CE1);
    recip_k<<<nb(CE2, T), T, 0, stream>>>(de2, CE2);
    recip_k<<<nb(CN1, T), T, 0, stream>>>(c0, CN1);
    recip_k<<<nb(CN2, T), T, 0, stream>>>(c1, CN2);

    auto gemm = [&](const float* A, const float* W, const float* b, float* Yd, int n, int k, int dout) {
        long tiles = (long)((n + 15) / 16) * (dout / 16);
        gemm_bias_wmma_k<<<(int)((tiles + 7) / 8), 256, 0, stream>>>(A, W, b, Yd, n, k, dout);
    };

    // templated lap over feature width D (256 or 64)
    auto lap256 = [&](const float* Xin, float* Ebuf, float* Yd,
                      const int* node, const int* edge, const float* dvi, const float* dei,
                      long ne, long nn, long nnz, int relu) {
        zero(Ebuf, ne * HID);
        lap_edge_k<HID><<<nb(nnz * (HID / 4), T), T, 0, stream>>>(Xin, node, edge, dvi, Ebuf, nnz);
        zero(Yd, nn * HID);
        lap_node_k<HID><<<nb(nnz * (HID / 4), T), T, 0, stream>>>(Ebuf, node, edge, dei, Yd, nnz);
        if (relu) scale_rows_k<HID, 1><<<nb(nn * (HID / 4), T), T, 0, stream>>>(Yd, dvi, nn);
        else      scale_rows_k<HID, 0><<<nb(nn * (HID / 4), T), T, 0, stream>>>(Yd, dvi, nn);
    };

    // ---- level 0: relu(lap(X@W0^T + b0)) ----
    gemm(X, W0, b0, big0, CN0, IN_DIM, HID);
    lap256(big0, big1, big2, node0, edge0, dv0, de0, CE0, CN0, CNNZ0, 1);

    // ---- pool0 -> (N1, HID) ----
    zero(P1, (long)CN1 * HID);
    pool_k<HID><<<nb((long)CN0 * (HID / 4), T), T, 0, stream>>>(big2, assign0, P1, CN0);
    scale_rows_k<HID, 0><<<nb((long)CN1 * (HID / 4), T), T, 0, stream>>>(P1, c0, CN1);

    // ---- level 1: h1 = relu(lap(P1@W1^T + b1)) ----
    gemm(P1, W1, b1, big0, CN1, HID, HID);
    lap256(big0, big1, H1, node1, edge1, dv1, de1, CE1, CN1, CNNZ1, 1);

    // ---- pool1 -> (N2, HID) ----
    zero(P2, (long)CN2 * HID);
    pool_k<HID><<<nb((long)CN1 * (HID / 4), T), T, 0, stream>>>(H1, assign1, P2, CN1);
    scale_rows_k<HID, 0><<<nb((long)CN2 * (HID / 4), T), T, 0, stream>>>(P2, c1, CN2);

    // ---- level 2 (bottleneck): relu(lap(P2@W2^T + b2)) ----
    gemm(P2, W2, b2, G2b, CN2, HID, HID);
    lap256(G2b, E2b, Y2, node2, edge2, dv2, de2, CE2, CN2, CNNZ2, 1);

    // ---- unpool + skip concat -> (N1, 2*HID) in big0 ----
    unpool_concat_k<HID><<<nb((long)CN1 * (HID / 4), T), T, 0, stream>>>(Y2, H1, assign1, big0, CN1);

    // ---- up path: relu(lap(U@W3^T + b3)) ----
    gemm(big0, W3, b3, big1, CN1, 2 * HID, HID);
    lap256(big1, big2, Y3, node1, edge1, dv1, de1, CE1, CN1, CNNZ1, 1);

    // ---- head: lap(Y3@W4^T + b4), no relu, write d_out ----
    gemm(Y3, W4, b4, big0, CN1, HID, OUT_DIM);
    {
        zero(big1, (long)CE1 * OUT_DIM);
        lap_edge_k<OUT_DIM><<<nb((long)CNNZ1 * (OUT_DIM / 4), T), T, 0, stream>>>(big0, node1, edge1, dv1, big1, CNNZ1);
        zero(out, (long)CN1 * OUT_DIM);
        lap_node_k<OUT_DIM><<<nb((long)CNNZ1 * (OUT_DIM / 4), T), T, 0, stream>>>(big1, node1, edge1, de1, out, CNNZ1);
        scale_rows_k<OUT_DIM, 0><<<nb((long)CN1 * (OUT_DIM / 4), T), T, 0, stream>>>(out, dv1, CN1);
    }
}
